// LightConvLayer_81939386073589
// MI455X (gfx1250) — matrix operand order repaired and checked by reading, not compiled, not run on hardware
//
#include <hip/hip_runtime.h>
#include <hip/hip_bf16.h>

// ---------------------------------------------------------------------------
// LightConv layer for MI455X (gfx1250, wave32):
//   y = (depthwise_conv3(x, softmax(w))) @ fc_w^T
// GEMM via V_WMMA_F32_16X16X32_BF16 with bf16 hi/lo split-precision
// (A_hi*B_hi + A_hi*B_lo + A_lo*B_hi) for ~fp32 accuracy at bf16 rate.
// x tile staged into LDS via the Tensor Data Mover (iterate mode gives the
// padded stride-516 row layout directly); falls back to manual staging if
// the TDM builtin is unavailable on this toolchain.
// ---------------------------------------------------------------------------

typedef __attribute__((ext_vector_type(16))) __bf16         v16bf;
typedef __attribute__((ext_vector_type(16))) unsigned short v16u;
typedef __attribute__((ext_vector_type(8)))  float          v8f;
typedef __attribute__((ext_vector_type(4)))  unsigned int   u32x4;
typedef __attribute__((ext_vector_type(8)))  int            i32x8;
typedef __attribute__((ext_vector_type(4)))  int            i32x4;

#define D_MODEL   512
#define M_TOTAL   65536          // 64*1024 rows
#define MT        128            // rows per block (8 waves x 16)
#define NT_BLK    128            // cols per block (8 tiles of 16)
#define KTILES    (D_MODEL / 32) // 16
#define NTILES    (D_MODEL / 16) // 32
#define ROWSTRIDE 516            // floats; 516*4 = 2064 ≡ 16 (mod 256) banks
#define SMEM_FLOATS ((MT + 1) * ROWSTRIDE)
#define SMEM_BYTES  (SMEM_FLOATS * 4)
#define BFRAG_ELEMS (KTILES * NTILES * 32 * 16)  // 262144 ushorts per array

#if defined(__gfx1250__) && __has_builtin(__builtin_amdgcn_tensor_load_to_lds)
#define HAVE_TDM 1
#else
#define HAVE_TDM 0
#endif

__device__ __forceinline__ unsigned short f2bf(float f) {
  unsigned u = __builtin_bit_cast(unsigned, f);
  u += 0x7fffu + ((u >> 16) & 1u);               // round-to-nearest-even
  return (unsigned short)(u >> 16);
}
__device__ __forceinline__ float bf2f(unsigned short h) {
  return __builtin_bit_cast(float, (unsigned)h << 16);
}

// ---------------------------------------------------------------------------
// Pre-swizzle fc_w into per-lane WMMA B-fragment layout (hi/lo bf16).
// B is 32x16 (KxN) per tile; lane = N%16 (+16 for the K+8 shift);
// lane<16 holds K=0..7,16..23 (elem h<8 -> K=h, h>=8 -> K=h+8), lane>=16 holds same K+8.
// frag index = ((kt*NTILES + nt)*32 + lane)*16 + h   -> 32B/lane contiguous.
// ---------------------------------------------------------------------------
__global__ void lightconv_prep_bfrag(const float* __restrict__ fc,
                                     unsigned short* __restrict__ bhi,
                                     unsigned short* __restrict__ blo) {
  int i = blockIdx.x * blockDim.x + threadIdx.x;   // 0 .. BFRAG_ELEMS-1
  int h    = i & 15;
  int lane = (i >> 4) & 31;
  int nt   = (i >> 9) & 31;
  int kt   = i >> 14;
  int K = kt * 32 + ((lane >> 4) << 3) + ((h < 8) ? h : 8 + h);
  int N = nt * 16 + (lane & 15);
  float v = fc[(size_t)N * D_MODEL + K];           // y = out @ fc^T -> B[K][N]=fc[N][K]
  unsigned short hi = f2bf(v);
  bhi[i] = hi;
  blo[i] = f2bf(v - bf2f(hi));
}

// ---------------------------------------------------------------------------
// Fused depthwise-conv + GEMM. Block: 256 threads (8 waves), 128M x 128N tile.
// ---------------------------------------------------------------------------
__global__ __launch_bounds__(256, 1)
void lightconv_gemm(const float* __restrict__ x,
                    const float* __restrict__ wraw,
                    const unsigned short* __restrict__ bhi,
                    const unsigned short* __restrict__ blo,
                    float* __restrict__ out) {
  extern __shared__ float smem[];                  // (MT+1) x ROWSTRIDE floats
  const int tid  = threadIdx.x;
  const int lane = tid & 31;
  const int wave = tid >> 5;
  const int colbase = blockIdx.x * NT_BLK;         // N-block fast-varying:
  const int rowbase = blockIdx.y * MT;             // 4 consumers of each x tile
                                                   // are launch-adjacent (L2 reuse)

#if HAVE_TDM
  // --- TDM: DMA x rows [rowbase, rowbase+MT) into LDS rows 1..MT.
  // Iterate mode: each iteration moves one 512-float row; LDS advances by
  // ROWSTRIDE floats per iteration => padded, bank-conflict-free layout.
  if (wave == 0) {
    unsigned lds_dst = (unsigned)__builtin_amdgcn_groupstaticsize()
                     + 1u * ROWSTRIDE * 4u;                      // row 1, bytes
    unsigned long long ga =
        (unsigned long long)(const void*)(x + (size_t)rowbase * D_MODEL);

    u32x4 g0;
    g0[0] = 1u;                                    // count=1, user descriptor
    g0[1] = lds_dst;                               // lds_addr
    g0[2] = (unsigned)(ga & 0xffffffffu);          // global_addr[31:0]
    g0[3] = (unsigned)((ga >> 32) & 0x01ffffffu)   // global_addr[56:32]
          | (2u << 30);                            // type = 2 ("image")

    i32x8 g1;
    g1[0] = (int)((2u << 16) | (1u << 19));        // data_size=4B, iterate_en=1
    g1[1] = (int)((512u & 0xffffu) << 16);         // tensor_dim0[15:0]=512
    g1[2] = (int)(((unsigned)M_TOTAL & 0xffffu) << 16); // dim0 hi=0, dim1 lo
    g1[3] = (int)(((unsigned)M_TOTAL >> 16) & 0xffffu)  // tensor_dim1 hi
          | (int)(512u << 16);                     // tile_dim0 = 512
    g1[4] = 1;                                     // tile_dim1=1, tile_dim2=0
    g1[5] = 512;                                   // tensor_dim0_stride lo
    g1[6] = (int)((512u & 0xffffu) << 16);         // stride0 hi=0, stride1 lo
    g1[7] = 0;                                     // tensor_dim1_stride hi

    i32x4 g2;
    g2[0] = 1;                                     // tensor_dim2 (unused)
    g2[1] = ROWSTRIDE;                             // lds_addr_increment / iter
    g2[2] = D_MODEL;                               // global_addr_increment lo
    g2[3] = (int)((unsigned)(MT - 1) << 16);       // iterate_count = 127
    i32x4 g3 = {0, 0, 0, 0};

#if __clang_major__ >= 23
    i32x8 g4 = {0, 0, 0, 0, 0, 0, 0, 0};
    __builtin_amdgcn_tensor_load_to_lds(g0, g1, g2, g3, g4, 0);
#else
    __builtin_amdgcn_tensor_load_to_lds(g0, g1, g2, g3, 0);
#endif
    __builtin_amdgcn_s_wait_tensorcnt(0);
  }
#else
  // --- manual staging fallback ---
  for (int i = tid; i < MT * 32; i += 256) {       // 32 x float4 chunks per row
    int r = i >> 5, c = i & 31;
    const float4 v = *(const float4*)(x + (size_t)(rowbase + r) * D_MODEL + c * 4);
    *(float4*)(smem + (size_t)(r + 1) * ROWSTRIDE + c * 4) = v;
  }
#endif

  // zero pads: each row's floats [512..515] = 0. Row r's "x[-1]" reads row
  // (r-1) elem 515 == 0; "x[512]" reads own elem 512 == 0. (Disjoint from
  // the TDM-written regions, so no LDS write race.)
  for (int r = tid; r <= MT; r += 256) {
    float4 z = {0.f, 0.f, 0.f, 0.f};
    *(float4*)(smem + (size_t)r * ROWSTRIDE + 512) = z;
  }
  __syncthreads();

  // --- per-lane softmax taps: head = global_row % 8 == lane % 8 (MT,16 ≡ 0 mod 8) ---
  const int head = lane & 7;
  float t0 = wraw[head * 3 + 0], t1 = wraw[head * 3 + 1], t2 = wraw[head * 3 + 2];
  float mx = fmaxf(t0, fmaxf(t1, t2));
  float e0 = __expf(t0 - mx), e1 = __expf(t1 - mx), e2 = __expf(t2 - mx);
  float inv = 1.0f / (e0 + e1 + e2);
  const float w0 = e0 * inv, w1 = e1 * inv, w2 = e2 * inv;

  // A-frag source row in LDS (A layout: M = lane%16; lane>=16 takes K+8 shift)
  const float* myrow = smem + (size_t)(wave * 16 + (lane & 15) + 1) * ROWSTRIDE;
  const int kshift = (lane >> 4) << 3;             // 0 or 8

  v8f acc[8];
  #pragma unroll
  for (int n = 0; n < 8; ++n) acc[n] = v8f{};

  for (int kt = 0; kt < KTILES; ++kt) {
    // prefetch next K-tile's B fragments into cache
    if (kt + 1 < KTILES) {
      size_t pf = ((size_t)((kt + 1) * NTILES + (colbase >> 4)) * 32 + lane) * 16;
      __builtin_prefetch(bhi + pf, 0, 1);
      __builtin_prefetch(blo + pf, 0, 1);
    }

    // --- build A fragment: conv3 on two runs of 8 contiguous K values ---
    // lane<16: elems 0..7 -> K = kt*32+0..7,   elems 8..15 -> K = kt*32+16..23
    // lane>=16: same +8. Each run needs x[base-1 .. base+8].
    // Cheap split: hi = trunc16(v); rem = v - hi (exact); lo = trunc16(rem).
    // Error ~2^-16 rel, same order as the dropped A_lo*B_lo term.
    v16u ah, al;
    const int base0 = kt * 32 + kshift;
    #pragma unroll
    for (int run = 0; run < 2; ++run) {
      const int base = base0 + run * 16;
      float xm = myrow[base - 1];
      float4 xa = *(const float4*)(myrow + base);
      float4 xb = *(const float4*)(myrow + base + 4);
      float xp = myrow[base + 8];
      float xv[10] = {xm, xa.x, xa.y, xa.z, xa.w, xb.x, xb.y, xb.z, xb.w, xp};
      #pragma unroll
      for (int h = 0; h < 8; ++h) {
        float cv = fmaf(w0, xv[h], fmaf(w1, xv[h + 1], w2 * xv[h + 2]));
        unsigned u = __builtin_bit_cast(unsigned, cv);
        ah[run * 8 + h] = (unsigned short)(u >> 16);
        float rem = cv - __builtin_bit_cast(float, u & 0xffff0000u);
        al[run * 8 + h] =
            (unsigned short)(__builtin_bit_cast(unsigned, rem) >> 16);
      }
    }
    const v16bf A_hi = __builtin_bit_cast(v16bf, ah);
    const v16bf A_lo = __builtin_bit_cast(v16bf, al);

    // --- 8 N-tiles: load B hi/lo frags (32B/lane coalesced), 3 WMMAs each ---
    #pragma unroll
    for (int nt = 0; nt < 8; ++nt) {
      const int ntg = (colbase >> 4) + nt;
      const size_t fidx = ((size_t)(kt * NTILES + ntg) * 32 + lane) * 16;
      const v16bf Bh = __builtin_bit_cast(v16bf, *(const v16u*)(bhi + fidx));
      const v16bf Bl = __builtin_bit_cast(v16bf, *(const v16u*)(blo + fidx));
      acc[nt] = __builtin_amdgcn_wmma_f32_16x16x32_bf16(
          false, A_hi, false, Bh, (short)0, acc[nt], false, false);
      acc[nt] = __builtin_amdgcn_wmma_f32_16x16x32_bf16(
          false, A_hi, false, Bl, (short)0, acc[nt], false, false);
      acc[nt] = __builtin_amdgcn_wmma_f32_16x16x32_bf16(
          false, A_lo, false, Bh, (short)0, acc[nt], false, false);
    }
  }

  // --- store: D layout VGPR v -> M = v (+8 for lanes 16..31), N = lane%16 ---
  const int mrow  = rowbase + wave * 16 + ((lane >= 16) ? 8 : 0);
  const int ncol0 = colbase + (lane & 15);
  #pragma unroll
  for (int nt = 0; nt < 8; ++nt) {
    #pragma unroll
    for (int v = 0; v < 8; ++v) {
      out[(size_t)(mrow + v) * D_MODEL + ncol0 + nt * 16] = acc[nt][v];
    }
  }
}

// ---------------------------------------------------------------------------
extern "C" void kernel_launch(void* const* d_in, const int* in_sizes, int n_in,
                              void* d_out, int out_size, void* d_ws, size_t ws_size,
                              hipStream_t stream) {
  const float* x  = (const float*)d_in[0];   // [64,1024,512] f32
  const float* wt = (const float*)d_in[1];   // [8,1,3] f32
  const float* fc = (const float*)d_in[2];   // [512,512] f32
  float* out = (float*)d_out;                // [64,1024,512] f32

  unsigned short* bhi = (unsigned short*)d_ws;
  unsigned short* blo = bhi + BFRAG_ELEMS;   // total ws use: 1 MiB

  // allow >64KB dynamic LDS (WGP has 320KB)
  hipFuncSetAttribute(reinterpret_cast<const void*>(lightconv_gemm),
                      hipFuncAttributeMaxDynamicSharedMemorySize, SMEM_BYTES);

  lightconv_prep_bfrag<<<BFRAG_ELEMS / 256, 256, 0, stream>>>(fc, bhi, blo);

  dim3 grid(D_MODEL / NT_BLK, M_TOTAL / MT);   // (4, 512): N-block fast
  lightconv_gemm<<<grid, 256, SMEM_BYTES, stream>>>(x, wt, bhi, blo, out);
}
